// GAT_63677185130602
// MI455X (gfx1250) — compile-verified, matrix operand order checked by reading, not run on hardware
//
#include <hip/hip_runtime.h>
#include <hip/hip_fp16.h>

typedef __attribute__((ext_vector_type(16))) _Float16 v16h;
typedef __attribute__((ext_vector_type(8)))  _Float16 v8h;
typedef __attribute__((ext_vector_type(8)))  float    v8f;

#define NN 50000
#define NE 800000
#define IN_DIM 128
#define HD 64           // hidden dim per head (== OUT dim per head)
#define H 4             // heads
#define FE 256          // H*HD = feature width of both GEMM outputs

// ---------------- utility kernels ----------------

__global__ void k_cast_f16(const float* __restrict__ in, _Float16* __restrict__ out, int n) {
  int i = blockIdx.x * blockDim.x + threadIdx.x;
  if (i < n) out[i] = (_Float16)in[i];
}

__global__ void k_fill_u32(unsigned* __restrict__ p, unsigned v, int n) {
  int i = blockIdx.x * blockDim.x + threadIdx.x;
  if (i < n) p[i] = v;
}

// Pre-swizzle B [K,N] f32 row-major into per-lane WMMA fragment order:
// Bs[((nt*(K/32)+kc)*32 + lane)*16 + p] so the GEMM reads one contiguous
// 32-byte vector per lane per K-chunk (2x global_load_b128, no stitching).
__global__ void k_swz_b(const float* __restrict__ W, _Float16* __restrict__ Bs,
                        int N, int K) {
  int t = blockIdx.x * blockDim.x + threadIdx.x;
  if (t >= N * K) return;
  const int KC = K >> 5;
  const int p    = t & 15;
  const int lane = (t >> 4) & 31;
  const int rest = t >> 9;          // nt*KC + kc
  const int kc = rest % KC;
  const int nt = rest / KC;
  const int half = lane >> 4, l15 = lane & 15;
  const int j = p >> 1, odd = p & 1;
  // 16-bit fragment layout (ISA 7.12.2): pairs j<4 -> K=2j, else K=16+2(j-4);
  // lanes 16..31 add +8.
  const int kk = kc * 32 + ((j < 4) ? 2 * j : 16 + 2 * (j - 4)) + 8 * half + odd;
  const int n = nt * 16 + l15;
  Bs[t] = (_Float16)W[(size_t)kk * N + n];
}

// ---------------- WMMA GEMM: C[M,N] = A[M,K] * Bswz, f16 in, f32 out ----
// one wave = one 16x64 output strip (4 WMMA tiles); 4 waves cover N=256.
__global__ __launch_bounds__(128)
void k_wmma_gemm(const _Float16* __restrict__ A, const _Float16* __restrict__ Bs,
                 float* __restrict__ C, int N, int K) {
  const int wave = threadIdx.x >> 5;
  const int lane = threadIdx.x & 31;
  const int half = lane >> 4;
  const int l15  = lane & 15;
  const int mt   = blockIdx.x;
  const int KC   = K >> 5;

  // A row for this lane; lanes 16..31 read the +8 K offset (fragment layout)
  const _Float16* arow = A + (size_t)(mt * 16 + l15) * K + 8 * half;

  v8f c[4] = {};
  for (int kc = 0; kc < KC; ++kc) {
    const v8h alo = *(const v8h*)(arow + kc * 32);        // K = 0..7  (+8*half)
    const v8h ahi = *(const v8h*)(arow + kc * 32 + 16);   // K = 16..23 (+8*half)
    const v16h a = __builtin_shufflevector(alo, ahi, 0, 1, 2, 3, 4, 5, 6, 7,
                                           8, 9, 10, 11, 12, 13, 14, 15);
#pragma unroll
    for (int t = 0; t < 4; ++t) {
      const int nt = wave * 4 + t;
      const v16h b = *(const v16h*)(Bs + ((size_t)(nt * KC + kc) * 32 + lane) * 16);
      c[t] = __builtin_amdgcn_wmma_f32_16x16x32_f16(false, a, false, b, (short)0,
                                                    c[t], false, false);
    }
  }
#pragma unroll
  for (int t = 0; t < 4; ++t) {
    const int ncol = (wave * 4 + t) * 16 + l15;
#pragma unroll
    for (int i = 0; i < 8; ++i) {
      const int m = mt * 16 + 8 * half + i;   // C/D: VGPR i -> M = i + 8*half
      C[(size_t)m * N + ncol] = c[t][i];
    }
  }
}

// ---------------- attention dot products: el/er [N,H] ----------------
__global__ void k_attn_dots(const float* __restrict__ feat, const float* __restrict__ al,
                            const float* __restrict__ ar, float* __restrict__ el,
                            float* __restrict__ er, int nh) {
  int i = blockIdx.x * blockDim.x + threadIdx.x;  // i = n*H + h
  if (i >= nh) return;
  const int h = i & (H - 1);
  const float* f = feat + (size_t)i * HD;         // [N,H,HD] contiguous
  const float* L = al + h * HD;
  const float* R = ar + h * HD;
  float sl = 0.f, sr = 0.f;
#pragma unroll 4
  for (int d = 0; d < HD; ++d) { float v = f[d]; sl += v * L[d]; sr += v * R[d]; }
  el[i] = sl;
  er[i] = sr;
}

// ---------------- edge pass 1: logits + segment max ----------------
__device__ __forceinline__ unsigned f32_ord(float f) {
  unsigned u = __float_as_uint(f);
  return (u & 0x80000000u) ? ~u : (u | 0x80000000u);
}

__global__ void k_edge_logits(const float* __restrict__ el, const float* __restrict__ er,
                              const int* __restrict__ src, const int* __restrict__ dst,
                              float* __restrict__ eb, unsigned* __restrict__ emax) {
  int i = blockIdx.x * blockDim.x + threadIdx.x;  // i = e*H + h
  if (i >= NE * H) return;
  const int e = i >> 2, h = i & 3;
  float v = el[src[e] * H + h] + er[dst[e] * H + h];
  v = v > 0.f ? v : 0.2f * v;                     // leaky_relu(0.2)
  eb[i] = v;
  atomicMax(&emax[dst[e] * H + h], f32_ord(v));
}

// decode ordered-uint max in place; empty segments -> 0 (matches reference)
__global__ void k_max_fix(unsigned* __restrict__ emax, int n) {
  int i = blockIdx.x * blockDim.x + threadIdx.x;
  if (i >= n) return;
  const unsigned u = emax[i];
  float f;
  if (u == 0x007FFFFFu) f = 0.f;                  // still at enc(-inf): no edges
  else f = (u & 0x80000000u) ? __uint_as_float(u ^ 0x80000000u) : __uint_as_float(~u);
  ((float*)emax)[i] = f;
}

// ---------------- edge pass 2: exp + segment sum ----------------
__global__ void k_edge_exp(float* __restrict__ eb, const float* __restrict__ emaxf,
                           float* __restrict__ denom, const int* __restrict__ dst) {
  int i = blockIdx.x * blockDim.x + threadIdx.x;
  if (i >= NE * H) return;
  const int e = i >> 2, h = i & 3;
  const int j = dst[e] * H + h;
  const float a = __expf(eb[i] - emaxf[j]);
  eb[i] = a;
  atomicAdd(&denom[j], a);
}

// ---------------- edge pass 3: alpha * feat[src] scattered to dst ----------
// 128 threads per edge (4 heads x 32 float2 pairs); a wave shares one (e,h).
__global__ void k_edge_scatter(const float* __restrict__ eb, const float* __restrict__ denom,
                               const float* __restrict__ feat, const int* __restrict__ src,
                               const int* __restrict__ dst, float* __restrict__ out,
                               int outStride, int headStride, float scale) {
  int t = blockIdx.x * blockDim.x + threadIdx.x;
  if (t >= NE * 128) return;
  const int e = t >> 7;
  const int r = t & 127;
  const int h = r >> 5;
  const int dp = (r & 31) * 2;
  const int s = src[e], d = dst[e];
  const float a  = eb[e * H + h];                 // wave-uniform broadcast load
  const float dn = denom[d * H + h];
  const float alpha = scale * a / fmaxf(dn, 1e-9f);
  const float* f = feat + (size_t)s * FE + h * HD + dp;
  float* o = out + (size_t)d * outStride + h * headStride + dp;
  atomicAdd(&o[0], f[0] * alpha);
  atomicAdd(&o[1], f[1] * alpha);
}

// ---------------- layer-1 epilogue: +bias, ELU, cast to f16 ----------------
__global__ void k_finish1(const float* __restrict__ acc, const float* __restrict__ b1,
                          _Float16* __restrict__ h16, int n) {
  int i = blockIdx.x * blockDim.x + threadIdx.x;
  if (i >= n) return;
  float v = acc[i] + b1[i & (FE - 1)];
  v = v > 0.f ? v : expm1f(v);                    // ELU(alpha=1)
  h16[i] = (_Float16)v;
}

// ---------------- layer-2 epilogue: add head-mean bias ----------------
__global__ void k_finish2(float* __restrict__ out, const float* __restrict__ b2, int n) {
  int i = blockIdx.x * blockDim.x + threadIdx.x;
  if (i >= n) return;
  const int d = i & (HD - 1);
  out[i] += 0.25f * (b2[d] + b2[HD + d] + b2[2 * HD + d] + b2[3 * HD + d]);
}

// ---------------- host launch ----------------
static inline int cdiv(int a, int b) { return (a + b - 1) / b; }

extern "C" void kernel_launch(void* const* d_in, const int* in_sizes, int n_in,
                              void* d_out, int out_size, void* d_ws, size_t ws_size,
                              hipStream_t stream) {
  const float* features = (const float*)d_in[0];
  const int*   src      = (const int*)d_in[1];
  const int*   dst      = (const int*)d_in[2];
  const float* W1 = (const float*)d_in[3];
  const float* al1 = (const float*)d_in[4];
  const float* ar1 = (const float*)d_in[5];
  const float* b1  = (const float*)d_in[6];
  const float* W2 = (const float*)d_in[7];
  const float* al2 = (const float*)d_in[8];
  const float* ar2 = (const float*)d_in[9];
  const float* b2  = (const float*)d_in[10];
  float* out = (float*)d_out;

  char* ws = (char*)d_ws;
  size_t off = 0;
  auto alloc = [&](size_t bytes) -> void* {
    void* p = ws + off;
    off = (off + bytes + 255) & ~(size_t)255;
    return p;
  };

  float*     feat  = (float*)alloc((size_t)NN * FE * 4);   // GEMM out (L1 then L2)
  float*     acc1  = (float*)alloc((size_t)NN * FE * 4);   // L1 aggregation
  _Float16*  xr    = (_Float16*)alloc((size_t)NN * FE * 2);// X-f16 then h-f16 (reused)
  _Float16*  W1s   = (_Float16*)alloc((size_t)IN_DIM * FE * 2);  // swizzled
  _Float16*  W2s   = (_Float16*)alloc((size_t)FE * FE * 2);      // swizzled
  float*     el    = (float*)alloc((size_t)NN * H * 4);
  float*     er    = (float*)alloc((size_t)NN * H * 4);
  unsigned*  emax  = (unsigned*)alloc((size_t)NN * H * 4);
  float*     denom = (float*)alloc((size_t)NN * H * 4);
  float*     eb    = (float*)alloc((size_t)NE * H * 4);

  const int NH = NN * H;       // 200000
  const int EH = NE * H;       // 3.2M
  const int NF = NN * FE;      // 12.8M
  const int NO = NN * HD;      // 3.2M  (final output elems)

  // ---- init ----
  k_fill_u32<<<cdiv(NO, 256), 256, 0, stream>>>((unsigned*)out, 0u, NO);
  k_cast_f16<<<cdiv(NN * IN_DIM, 256), 256, 0, stream>>>(features, xr, NN * IN_DIM);
  k_swz_b<<<cdiv(IN_DIM * FE, 256), 256, 0, stream>>>(W1, W1s, FE, IN_DIM);
  k_swz_b<<<cdiv(FE * FE, 256), 256, 0, stream>>>(W2, W2s, FE, FE);

  // ---- layer 1 ----
  k_wmma_gemm<<<NN / 16, 128, 0, stream>>>(xr, W1s, feat, FE, IN_DIM);
  k_attn_dots<<<cdiv(NH, 256), 256, 0, stream>>>(feat, al1, ar1, el, er, NH);
  k_fill_u32<<<cdiv(NH, 256), 256, 0, stream>>>(emax, 0x007FFFFFu, NH);      // enc(-inf)
  k_fill_u32<<<cdiv(NH, 256), 256, 0, stream>>>((unsigned*)denom, 0u, NH);
  k_fill_u32<<<cdiv(NF, 256), 256, 0, stream>>>((unsigned*)acc1, 0u, NF);
  k_edge_logits<<<cdiv(EH, 256), 256, 0, stream>>>(el, er, src, dst, eb, emax);
  k_max_fix<<<cdiv(NH, 256), 256, 0, stream>>>(emax, NH);
  k_edge_exp<<<cdiv(EH, 256), 256, 0, stream>>>(eb, (const float*)emax, denom, dst);
  k_edge_scatter<<<cdiv(NE * 128, 256), 256, 0, stream>>>(eb, denom, feat, src, dst,
                                                          acc1, FE, HD, 1.0f);
  k_finish1<<<cdiv(NF, 256), 256, 0, stream>>>(acc1, b1, xr, NF);  // xr now holds h16

  // ---- layer 2 ----
  k_wmma_gemm<<<NN / 16, 128, 0, stream>>>(xr, W2s, feat, FE, FE);
  k_attn_dots<<<cdiv(NH, 256), 256, 0, stream>>>(feat, al2, ar2, el, er, NH);
  k_fill_u32<<<cdiv(NH, 256), 256, 0, stream>>>(emax, 0x007FFFFFu, NH);
  k_fill_u32<<<cdiv(NH, 256), 256, 0, stream>>>((unsigned*)denom, 0u, NH);
  k_edge_logits<<<cdiv(EH, 256), 256, 0, stream>>>(el, er, src, dst, eb, emax);
  k_max_fix<<<cdiv(NH, 256), 256, 0, stream>>>(emax, NH);
  k_edge_exp<<<cdiv(EH, 256), 256, 0, stream>>>(eb, (const float*)emax, denom, dst);
  // heads collapse into d_out with 0.25 scale -> mean over heads for free
  k_edge_scatter<<<cdiv(NE * 128, 256), 256, 0, stream>>>(eb, denom, feat, src, dst,
                                                          out, HD, 0, 0.25f);
  k_finish2<<<cdiv(NO, 256), 256, 0, stream>>>(out, b2, NO);
}
